// ModulatedConv2d_61675730371262
// MI455X (gfx1250) — compile-verified, matrix operand order checked by reading, not run on hardware
//
#include <hip/hip_runtime.h>

typedef __attribute__((ext_vector_type(16))) _Float16 v16h;
typedef __attribute__((ext_vector_type(2)))  _Float16 v2h;
typedef __attribute__((ext_vector_type(8)))  float    v8f;

// Problem constants (fixed by the reference)
constexpr int BATCH = 4;
constexpr int CIN   = 512;
constexpr int COUT  = 512;
constexpr int SDIM  = 512;
constexpr int H     = 64;
constexpr int W     = 64;
constexpr int NPIX  = H * W;            // 4096 (GEMM N per batch)
constexpr int KSTEP = CIN * 9 / 32;     // 144 wmma K-steps (tap-major: kt = tap*16 + ib)
constexpr int MTILES = COUT / 16;       // 32
constexpr int NTILES = NPIX / 16;       // 256 (16 | 64 -> tile never crosses a row)
constexpr int MWAVE  = 8;               // M-tiles per wave (reuse B fragment 8x)
constexpr int MBLOCKS = MTILES / MWAVE; // 4

// Zero-padded activation plane: 66x66 (1-pixel halo) -> no im2col bounds checks.
// Layout: channel-PAIR interleaved: xh[b][i/2][padded_pixel][2], so one b32 load
// yields the two adjacent-channel halves of each WMMA B-fragment VGPR.
constexpr int PW    = W + 2;            // 66
constexpr int PLANE = PW * (H + 2);     // 4356 padded pixels per plane

constexpr float SGAIN = 0.04419417382415922f;   // 1/sqrt(512)
constexpr float WGAIN = 0.014731391274719739f;  // 1/sqrt(512*9)

// ---------------------------------------------------------------------------
// Kernel 1: per channel pair (2*i2, 2*i2+1):
//   s[b,i] = dot(style[b], affine_w[i]) * SGAIN + affine_b[i]
//   xh[b][i2][pix] = { f16(x[b,2i2,pix]*wg*s0), f16(x[b,2i2+1,pix]*wg*s1) }
// One block per (i2, b). 256 threads.
// ---------------------------------------------------------------------------
__global__ __launch_bounds__(256)
void modulate_pad_f16(const float* __restrict__ x,
                      const float* __restrict__ style,
                      const float* __restrict__ aff_w,
                      const float* __restrict__ aff_b,
                      _Float16* __restrict__ xh) {
  const int i2 = blockIdx.x;  // channel pair index
  const int b  = blockIdx.y;  // batch
  __shared__ float red0[256];
  __shared__ float red1[256];

  float p0 = 0.f, p1 = 0.f;
  const float* sp = style + (size_t)b * SDIM;
  const float* a0 = aff_w + (size_t)(2 * i2) * SDIM;
  const float* a1 = a0 + SDIM;
  for (int t = threadIdx.x; t < SDIM; t += 256) {
    const float sv = sp[t];
    p0 += sv * a0[t];
    p1 += sv * a1[t];
  }
  red0[threadIdx.x] = p0;
  red1[threadIdx.x] = p1;
  __syncthreads();
  #pragma unroll
  for (int off = 128; off > 0; off >>= 1) {
    if ((int)threadIdx.x < off) {
      red0[threadIdx.x] += red0[threadIdx.x + off];
      red1[threadIdx.x] += red1[threadIdx.x + off];
    }
    __syncthreads();
  }
  const float scale0 = (red0[0] * SGAIN + aff_b[2 * i2])     * WGAIN;
  const float scale1 = (red1[0] * SGAIN + aff_b[2 * i2 + 1]) * WGAIN;

  const float* xp0 = x + ((size_t)b * CIN + 2 * i2) * NPIX;
  const float* xp1 = xp0 + NPIX;
  v2h* op = (v2h*)(xh + ((size_t)b * CIN + 2 * (size_t)i2) * PLANE);
  for (int t = threadIdx.x; t < PLANE; t += 256) {
    const int r = t / PW;
    const int c = t - r * PW;
    const bool in = (r >= 1) & (r <= H) & (c >= 1) & (c <= W);
    const int idx = (r - 1) * W + (c - 1);
    v2h pr;
    pr.x = in ? (_Float16)(xp0[idx] * scale0) : (_Float16)0.f;
    pr.y = in ? (_Float16)(xp1[idx] * scale1) : (_Float16)0.f;
    op[t] = pr;
  }
}

// ---------------------------------------------------------------------------
// Kernel 2: pack fp32 weight [O, I, 3, 3] into f16 A-fragment layout, K order
// tap-major:  k = tap*512 + i,  kt = tap*16 + ib.
//   apack[((kt*32 + mTile)*32 + lane)*16 + j]
// Fragment element j of lane L holds A[m, k]:
//   m = mTile*16 + (L & 15)
//   k_local = (j < 8 ? j : j + 8) + ((L >> 4) << 3)   (CDNA5 16-bit A layout)
// ---------------------------------------------------------------------------
__global__ __launch_bounds__(256)
void pack_weight(const float* __restrict__ w, _Float16* __restrict__ apack) {
  const int gid = blockIdx.x * 256 + threadIdx.x;   // one per (kt, mTile, lane)
  if (gid >= KSTEP * MTILES * 32) return;
  const int lane  = gid & 31;
  const int mTile = (gid >> 5) & 31;
  const int kt    = gid >> 10;                      // 0..143
  const int m     = mTile * 16 + (lane & 15);
  const int kgrp  = (lane >> 4) << 3;
  const int tap   = kt >> 4;                        // kh*3 + kw
  const int ibase = (kt & 15) * 32;

  _Float16* dst = apack + (size_t)gid * 16;
  #pragma unroll
  for (int j = 0; j < 16; ++j) {
    const int kl = (j < 8 ? j : j + 8) + kgrp;
    const int i  = ibase + kl;
    dst[j] = (_Float16)w[((size_t)m * CIN + i) * 9 + tap];
  }
}

// ---------------------------------------------------------------------------
// Kernel 3: implicit GEMM.  Out[b][o][n] = sum_k A[o][k] * B_im2col[k][n]
// 8 waves / block; each wave: one 16-pixel N-tile, 8 stacked M-tiles (128 rows).
// B fragment = 8 aligned b32 loads (channel-pair layout), A = b128 clauses.
// ---------------------------------------------------------------------------
__global__ __launch_bounds__(256)
void mconv_gemm(const _Float16* __restrict__ xh,
                const _Float16* __restrict__ apack,
                float* __restrict__ out) {
  const int wave = threadIdx.x >> 5;
  const int lane = threadIdx.x & 31;
  const int job  = blockIdx.x * 8 + wave;    // 0 .. BATCH*MBLOCKS*NTILES-1

  const int b      = job >> 10;              // / (MBLOCKS * NTILES) = 1024
  const int rem    = job & 1023;
  const int mblock = rem >> 8;               // 0..3
  const int nTile  = rem & 255;

  const int hrow  = nTile >> 2;              // image row of this tile
  const int wbase = (nTile & 3) << 4;        // first pixel column
  const int laneN = lane & 15;
  const int kgrp  = (lane >> 4) << 3;        // +8 channels = +4 pair-planes

  // B base: batch + lane-group channel offset (kgrp*PLANE elems) + pixel column.
  // Pair layout: addr(elem) = pairPlane*2*PLANE + pixel*2  (+0/1 channel parity)
  const _Float16* xw = xh + ((size_t)b * CIN + kgrp) * PLANE
                          + 2 * (size_t)(wbase + laneN);
  // A pointer marches linearly: apack chunk per kt is 32*32*16 = 16384 elems.
  const _Float16* aptr = apack + ((size_t)(mblock * MWAVE) * 32 + lane) * 16;

  v8f acc[MWAVE];
  #pragma unroll
  for (int mt = 0; mt < MWAVE; ++mt)
    acc[mt] = (v8f){0.f,0.f,0.f,0.f,0.f,0.f,0.f,0.f};

  #pragma unroll
  for (int kh = 0; kh < 3; ++kh) {
    #pragma unroll
    for (int kw = 0; kw < 3; ++kw) {
      const _Float16* bptr = xw + 2 * ((hrow + kh) * PW + kw);
      for (int ib = 0; ib < 16; ++ib) {        // 16 channel-blocks of 32
        __builtin_prefetch(bptr + 32 * PLANE, 0, 1);
        // B fragment: VGPR v holds k_local pair (2v, 2v+1) (+8 for v>=4)
        union { v16h h; int w32[8]; } bu;
        #pragma unroll
        for (int v = 0; v < 8; ++v) {
          const int pv = (v < 4) ? v : v + 4;  // pair-plane offset 0..3, 8..11
          bu.w32[v] = *(const int*)(bptr + (size_t)pv * (2 * PLANE));
        }
        #pragma unroll
        for (int mt = 0; mt < MWAVE; ++mt) {
          const v16h af = *(const v16h*)(aptr + mt * 512);  // imm offset mt*1024B
          acc[mt] = __builtin_amdgcn_wmma_f32_16x16x32_f16(
              false, af, false, bu.h, (short)0, acc[mt], false, false);
        }
        bptr += 32 * PLANE;                    // next 32 channels (16 pair-planes)
        aptr += MTILES * 32 * 16;              // next kt chunk (32 KB)
      }
    }
  }

  // Store D: VGPR r, lane L -> M = r + 8*(L>=16), N = L&15
  const int mhi = (lane >> 4) << 3;
  float* obase = out + ((size_t)b * COUT + mblock * (MWAVE * 16) + mhi) * NPIX
                     + hrow * W + wbase + laneN;
  #pragma unroll
  for (int mt = 0; mt < MWAVE; ++mt) {
    #pragma unroll
    for (int r = 0; r < 8; ++r)
      obase[(size_t)(mt * 16 + r) * NPIX] = acc[mt][r];   // imm offsets < 8MB
  }
}

// ---------------------------------------------------------------------------
extern "C" void kernel_launch(void* const* d_in, const int* in_sizes, int n_in,
                              void* d_out, int out_size, void* d_ws, size_t ws_size,
                              hipStream_t stream) {
  const float* x     = (const float*)d_in[0];   // [4,512,64,64]
  const float* style = (const float*)d_in[1];   // [4,512]
  const float* wgt   = (const float*)d_in[2];   // [512,512,3,3]
  const float* affw  = (const float*)d_in[3];   // [512,512]
  const float* affb  = (const float*)d_in[4];   // [512]
  float* out = (float*)d_out;                   // [4,512,64,64]

  // workspace: xh padded pair-planes (17.9 MB) | apack (4.7 MB)
  _Float16* xh    = (_Float16*)d_ws;
  size_t xh_bytes = (size_t)BATCH * CIN * PLANE * sizeof(_Float16);
  xh_bytes = (xh_bytes + 255) & ~(size_t)255;
  _Float16* apack = (_Float16*)((char*)d_ws + xh_bytes);

  modulate_pad_f16<<<dim3(CIN / 2, BATCH), 256, 0, stream>>>(x, style, affw, affb, xh);

  const int packJobs = KSTEP * MTILES * 32;     // 147456
  pack_weight<<<(packJobs + 255) / 256, 256, 0, stream>>>(wgt, apack);

  const int jobs   = BATCH * MBLOCKS * NTILES;  // 4096 wave-jobs
  const int blocks = jobs / 8;                  // 512 blocks of 8 waves
  mconv_gemm<<<blocks, 256, 0, stream>>>(xh, apack, out);
}
